// SupervisedGCNN_81174881894972
// MI455X (gfx1250) — compile-verified, matrix operand order checked by reading, not run on hardware
//
#include <hip/hip_runtime.h>
#include <hip/hip_bf16.h>

typedef __attribute__((ext_vector_type(16))) _Float16 v16h;
typedef __attribute__((ext_vector_type(8)))  float    v8f;

#define WAVE 32
#define BLK  256   // 8 waves per block

// ---------------- degree / normalization ----------------
__global__ void deg_init_kernel(float* deg, int n) {
    int i = blockIdx.x * blockDim.x + threadIdx.x;
    if (i < n) deg[i] = 1.0f;                    // self loop contributes 1
}

__global__ void deg_count_kernel(const long long* __restrict__ dst, float* deg, int e) {
    int i = blockIdx.x * blockDim.x + threadIdx.x;
    if (i < e) atomicAdd(&deg[(int)dst[i]], 1.0f);
}

__global__ void deg_rsqrt_kernel(float* deg, int n) {
    int i = blockIdx.x * blockDim.x + threadIdx.x;
    if (i < n) {
        float d = deg[i];
        deg[i] = (d > 0.0f) ? rsqrtf(d) : 0.0f;  // becomes dinv
    }
}

// ---------------- WMMA GEMM: C[M,N] = A[M,K] * B[K,N] ----------------
// Block = 8 waves. Block owns one 16-column B panel (tileN) staged once into
// LDS as f16, transposed [col][k] with padding. Each wave computes one 16x16
// C tile (8 consecutive M tiles per block) with v_wmma_f32_16x16x32_f16.
template <int K_DIM, int N_DIM>
__global__ __launch_bounds__(BLK) void gemm_wmma_kernel(
        const float* __restrict__ A,
        const float* __restrict__ B,
        float* __restrict__ C, int M) {
    __shared__ _Float16 bsh[16][K_DIM + 8];      // +8 halves: stagger LDS banks

    const int ntilesN = N_DIM / 16;
    const int tileN   = blockIdx.x % ntilesN;
    const int mblock  = blockIdx.x / ntilesN;
    const int wavein  = threadIdx.x / WAVE;      // uniform per wave
    const int lane    = threadIdx.x & (WAVE - 1);
    const int tileM   = mblock * 8 + wavein;

    // ---- cooperative B-panel staging: K_DIM x 16 cols, f32 -> f16, transposed
    const int col0 = tileN * 16;
    #pragma unroll
    for (int j = 0; j < K_DIM / 16; ++j) {       // 16*K_DIM elements / 256 threads
        int e = threadIdx.x + BLK * j;
        int k = e >> 4;
        int c = e & 15;
        bsh[c][k] = (_Float16)B[(size_t)k * N_DIM + col0 + c];
    }
    __syncthreads();

    if (tileM * 16 < M) {                        // wave-uniform: EXEC all-1s inside
        const int half = lane >> 4;              // 0: K 0-7/16-23, 1: K 8-15/24-31
        const int idx  = lane & 15;
        const float*    ap = A + (size_t)(tileM * 16 + idx) * K_DIM;  // A row per lane
        const _Float16* bp = &bsh[idx][0];                            // B col per lane

        v8f c = {};
        #pragma unroll
        for (int k0 = 0; k0 < K_DIM; k0 += 32) {
            int ka = k0 + half * 8;
            v16h a, b;
            #pragma unroll
            for (int i = 0; i < 8; ++i) {
                a[i]     = (_Float16)ap[ka + i];        // contiguous 32B -> 2x b128
                a[i + 8] = (_Float16)ap[ka + 16 + i];
            }
            #pragma unroll
            for (int i = 0; i < 8; ++i) {
                b[i]     = bp[ka + i];                  // contiguous 16B -> ds_load_b128
                b[i + 8] = bp[ka + 16 + i];
            }
            c = __builtin_amdgcn_wmma_f32_16x16x32_f16(
                    /*neg_a=*/false, a, /*neg_b=*/false, b,
                    /*c_mod=*/(short)0, c, /*reuse_a=*/false, /*reuse_b=*/false);
        }

        // C/D layout: lane idx = column, VGPR j -> row j (lanes 0-15) or j+8
        const int col = tileN * 16 + idx;
        #pragma unroll
        for (int j = 0; j < 8; ++j) {
            int row = tileM * 16 + j + half * 8;
            C[(size_t)row * N_DIM + col] = c[j];
        }
    }
}

// ---------------- aggregation ----------------
// out[i,c] = h[i,c] * dinv[i]^2 + bias[c]   (self-loop term; fully overwrites out)
template <int D>
__global__ void agg_init_kernel(const float* __restrict__ h,
                                const float* __restrict__ dinv,
                                const float* __restrict__ bias,
                                float* __restrict__ out, int n) {
    long long i = (long long)blockIdx.x * blockDim.x + threadIdx.x;
    long long total = (long long)n * D;
    if (i >= total) return;
    int node = (int)(i / D);
    int c    = (int)(i % D);
    float s = dinv[node];
    out[i] = h[i] * s * s + bias[c];
}

// per-edge scatter: out[dst] += h[src] * dinv[src]*dinv[dst]  (float4 vectorized)
template <int D>
__global__ void agg_edge_kernel(const float* __restrict__ h,
                                const long long* __restrict__ src,
                                const long long* __restrict__ dst,
                                const float* __restrict__ dinv,
                                float* __restrict__ out, int e) {
    constexpr int CH = D >> 2;                 // float4 chunks per edge (power of 2)
    long long t = (long long)blockIdx.x * blockDim.x + threadIdx.x;
    long long total = (long long)e * CH;
    if (t >= total) return;
    int edge = (int)(t / CH);
    int c4   = (int)(t % CH);
    int s  = (int)src[edge];
    int dd = (int)dst[edge];
    float nrm = dinv[s] * dinv[dd];
    const float4 v = *(const float4*)(h + (size_t)s * D + 4 * c4);
    float* o = out + (size_t)dd * D + 4 * c4;
    atomicAdd(o + 0, v.x * nrm);
    atomicAdd(o + 1, v.y * nrm);
    atomicAdd(o + 2, v.z * nrm);
    atomicAdd(o + 3, v.w * nrm);
}

__global__ void relu_kernel(float* buf, long long n) {
    long long i = (long long)blockIdx.x * blockDim.x + threadIdx.x;
    if (i < n) buf[i] = fmaxf(buf[i], 0.0f);
}

// ---------------- launch ----------------
extern "C" void kernel_launch(void* const* d_in, const int* in_sizes, int n_in,
                              void* d_out, int out_size, void* d_ws, size_t ws_size,
                              hipStream_t stream) {
    const float*     x   = (const float*)d_in[0];
    const long long* ei  = (const long long*)d_in[1];
    const float*     W1  = (const float*)d_in[2];
    const float*     b1  = (const float*)d_in[3];
    const float*     W2  = (const float*)d_in[4];
    const float*     b2  = (const float*)d_in[5];
    float*           out = (float*)d_out;

    const int IN = 256, HID = 256, OUT = 64;
    const int M = in_sizes[0] / IN;         // 50000 (multiple of 16)
    const int E = in_sizes[1] / 2;          // 800000
    const long long* src = ei;
    const long long* dst = ei + E;

    // workspace layout: dinv | buf1 (M*HID f32) | buf2 (M*HID f32)
    char* ws = (char*)d_ws;
    float* dinv = (float*)ws;
    size_t off = ((size_t)M * sizeof(float) + 255) & ~(size_t)255;
    float* buf1 = (float*)(ws + off);
    off += (((size_t)M * HID * sizeof(float)) + 255) & ~(size_t)255;
    float* buf2 = (float*)(ws + off);

    // degrees -> dinv
    deg_init_kernel <<<(M + BLK - 1) / BLK, BLK, 0, stream>>>(dinv, M);
    deg_count_kernel<<<(E + BLK - 1) / BLK, BLK, 0, stream>>>(dst, dinv, E);
    deg_rsqrt_kernel<<<(M + BLK - 1) / BLK, BLK, 0, stream>>>(dinv, M);

    const int mtiles  = M / 16;                    // 3125
    const int mblocks = (mtiles + 7) / 8;          // 8 M-tiles per block

    // layer 1: buf1 = x @ W1
    gemm_wmma_kernel<256, 256><<<mblocks * (HID / 16), BLK, 0, stream>>>(x, W1, buf1, M);

    // buf2 = D^-1/2 (A+I) D^-1/2 buf1 + b1 ; relu
    {
        long long total = (long long)M * HID;
        agg_init_kernel<256><<<(unsigned)((total + BLK - 1) / BLK), BLK, 0, stream>>>(buf1, dinv, b1, buf2, M);
        long long etot = (long long)E * (HID / 4);
        agg_edge_kernel<256><<<(unsigned)((etot + BLK - 1) / BLK), BLK, 0, stream>>>(buf1, src, dst, dinv, buf2, E);
        relu_kernel<<<(unsigned)((total + BLK - 1) / BLK), BLK, 0, stream>>>(buf2, total);
    }

    // layer 2: buf1 (reused, M*OUT) = buf2 @ W2
    gemm_wmma_kernel<256, 64><<<mblocks * (OUT / 16), BLK, 0, stream>>>(buf2, W2, buf1, M);

    // d_out = aggregate(buf1) + b2
    {
        long long total = (long long)M * OUT;
        agg_init_kernel<64><<<(unsigned)((total + BLK - 1) / BLK), BLK, 0, stream>>>(buf1, dinv, b2, out, M);
        long long etot = (long long)E * (OUT / 4);
        agg_edge_kernel<64><<<(unsigned)((etot + BLK - 1) / BLK), BLK, 0, stream>>>(buf1, src, dst, dinv, out, E);
    }
}
